// WindowAttention_50319836840386
// MI455X (gfx1250) — compile-verified, matrix operand order checked by reading, not run on hardware
//
#include <hip/hip_runtime.h>
#include <hip/hip_bf16.h>

typedef __attribute__((ext_vector_type(16))) __bf16 v16bf;
typedef __attribute__((ext_vector_type(8)))  __bf16 v8bf;
typedef __attribute__((ext_vector_type(8)))  float  v8f;
typedef __attribute__((ext_vector_type(4)))  float  v4f;

#define WMMA_BF16(a, b, c) \
    __builtin_amdgcn_wmma_f32_16x16x32_bf16(false, (a), false, (b), (short)0, (c), false, false)

// Load one 16x32 bf16 WMMA operand slice for this lane.
// Per ISA layout (16-bit A 16x32): lane half h reads K=[8h..8h+7] and K=[16+8h..16+8h+7]
// -> two contiguous 16-byte loads from the lane's row. Same pattern used for B
// (lane = output column, row = weight row, contiguous in memory).
__device__ __forceinline__ v16bf load_op(const __bf16* row, int h) {
    v16bf t;
    ((v8bf*)&t)[0] = *(const v8bf*)(row + 8 * h);
    ((v8bf*)&t)[1] = *(const v8bf*)(row + 16 + 8 * h);
    return t;
}

// ---- prep: cast weights f32 -> bf16 (L2-resident, reused by all 3136 blocks) ----
__global__ void prep_weights(const float* __restrict__ qkv_w, const float* __restrict__ proj_w,
                             __bf16* __restrict__ wq, __bf16* __restrict__ wp) {
    int i = blockIdx.x * blockDim.x + threadIdx.x;
    if (i < 1152 * 384) wq[i] = (__bf16)qkv_w[i];
    if (i < 384 * 384)  wp[i] = (__bf16)proj_w[i];
}

// LDS element-offset map (in __bf16 units), 256000 bytes total:
//   xA   : [0, 64*392)                     window activations / later attn-out, row stride 392
//   head : 25088 + head*8576, per head:
//     q  : +0     64 rows stride 40
//     k  : +2560  64 rows stride 40
//     vT : +5120  32 rows stride 72
//     P  : +7424  16 rows stride 72
#define LDS_BYTES 256000

__global__ __launch_bounds__(384, 1)
void win_attn_fused(const float* __restrict__ x,
                    const __bf16* __restrict__ wq,
                    const float* __restrict__ qkv_b,
                    const __bf16* __restrict__ wp,
                    const float* __restrict__ proj_b,
                    const float* __restrict__ bt,
                    float* __restrict__ out) {
    extern __shared__ __align__(16) __bf16 sh[];

    const int tid  = threadIdx.x;
    const int wave = tid >> 5;
    const int lane = tid & 31;
    const int idx  = lane & 15;   // column within 16-lane group
    const int h    = lane >> 4;   // lane half (K/M-row half select)

    const int wi  = blockIdx.x;           // window id 0..3135
    const int bb  = wi / 784;
    const int rem = wi - bb * 784;
    const int wy  = rem / 28;
    const int wx  = rem - wy * 28;
    // base pixel offset of this window; full addr = pixbase + c*50176 + ty*224 + tx
    const long pixbase = (long)bb * 384 * 50176 + (long)(wy * 8) * 224 + wx * 8;

    // ---------------- fill xA (window partition + f32->bf16) ----------------
    {
        int g  = tid / 64;        // channel sub-group 0..5
        int t  = tid & 63;        // token
        int ty = t >> 3, tx = t & 7;
        const float* px = x + pixbase + ty * 224 + tx;
        for (int cc = 0; cc < 384; cc += 6) {
            int c = cc + g;
            sh[t * 392 + c] = (__bf16)px[(long)c * 50176];
        }
    }
    __syncthreads();

    // ---------------- phase 1: QKV projection (M=64,K=384,N=1152) ----------------
    for (int i = 0; i < 6; ++i) {
        int ntg = wave * 6 + i;
        int o   = ntg * 16 + idx;                 // output feature (B column)
        const __bf16* wrow = wq + (long)o * 384;
        v8f acc[4];
        for (int m = 0; m < 4; ++m)
            for (int r = 0; r < 8; ++r) acc[m][r] = 0.f;
        for (int kc = 0; kc < 12; ++kc) {
            v16bf bop = load_op(wrow + kc * 32, h);
            for (int m = 0; m < 4; ++m) {
                v16bf aop = load_op(sh + (m * 16 + idx) * 392 + kc * 32, h);
                acc[m] = WMMA_BF16(aop, bop, acc[m]);
            }
        }
        float bias = qkv_b[o];
        int part = o / 384;                       // uniform per wave: 0=q,1=k,2=v
        int oc   = o - part * 384;
        int head = oc >> 5;
        int d    = oc & 31;
        __bf16* hb = sh + 25088 + head * 8576;
        if (part == 0) {
            const float scale = 0.17677669529663687f;   // 32^-0.5
            for (int m = 0; m < 4; ++m)
                for (int r = 0; r < 8; ++r) {
                    int t = m * 16 + 8 * h + r;
                    hb[t * 40 + d] = (__bf16)((acc[m][r] + bias) * scale);
                }
        } else if (part == 1) {
            for (int m = 0; m < 4; ++m)
                for (int r = 0; r < 8; ++r) {
                    int t = m * 16 + 8 * h + r;
                    hb[2560 + t * 40 + d] = (__bf16)(acc[m][r] + bias);
                }
        } else {
            // store v transposed: vT[d][t], 8 consecutive tokens -> one b128 store
            for (int m = 0; m < 4; ++m) {
                v8bf pk;
                for (int r = 0; r < 8; ++r) pk[r] = (__bf16)(acc[m][r] + bias);
                *(v8bf*)(hb + 5120 + d * 72 + m * 16 + 8 * h) = pk;
            }
        }
    }
    __syncthreads();

    // ---------------- phase 2: attention, wave = head ----------------
    {
        int head = wave;
        __bf16* hb = sh + 25088 + head * 8576;
        const __bf16* qb  = hb;
        const __bf16* kb  = hb + 2560;
        const __bf16* vtb = hb + 5120;
        __bf16*       Pb  = hb + 7424;

        v16bf kB[4], vB[2][2];
        for (int nt = 0; nt < 4; ++nt)
            kB[nt] = load_op(kb + (nt * 16 + idx) * 40, h);
        for (int dt = 0; dt < 2; ++dt)
            for (int kc = 0; kc < 2; ++kc)
                vB[dt][kc] = load_op(vtb + (dt * 16 + idx) * 72 + kc * 32, h);

        for (int mt = 0; mt < 4; ++mt) {
            v16bf qA = load_op(qb + (mt * 16 + idx) * 40, h);
            v8f s[4];
            for (int nt = 0; nt < 4; ++nt) {
                for (int r = 0; r < 8; ++r) s[nt][r] = 0.f;
                s[nt] = WMMA_BF16(qA, kB[nt], s[nt]);   // K=32 -> single WMMA
            }
            // relative position bias (arithmetic gather, table is L0-resident)
            for (int nt = 0; nt < 4; ++nt) {
                int j = nt * 16 + idx;
                int jy = j >> 3, jx = j & 7;
                for (int r = 0; r < 8; ++r) {
                    int ii = mt * 16 + 8 * h + r;
                    int iy = ii >> 3, ix = ii & 7;
                    int rel = (iy - jy + 7) * 15 + (ix - jx + 7);
                    s[nt][r] += bt[rel * 12 + head];
                }
            }
            // softmax over row (4 in-register cols * 16 lanes)
            float inv[8];
            for (int r = 0; r < 8; ++r) {
                float m0 = s[0][r];
                for (int nt = 1; nt < 4; ++nt) m0 = fmaxf(m0, s[nt][r]);
                for (int msk = 1; msk < 16; msk <<= 1) m0 = fmaxf(m0, __shfl_xor(m0, msk, 32));
                float a0 = 0.f;
                for (int nt = 0; nt < 4; ++nt) {
                    float p = __expf(s[nt][r] - m0);
                    s[nt][r] = p;
                    a0 += p;
                }
                for (int msk = 1; msk < 16; msk <<= 1) a0 += __shfl_xor(a0, msk, 32);
                inv[r] = 1.f / a0;
            }
            // write P (this wave's 16 rows) to LDS in A-operand row-major form
            for (int nt = 0; nt < 4; ++nt)
                for (int r = 0; r < 8; ++r)
                    Pb[(8 * h + r) * 72 + nt * 16 + idx] = (__bf16)(s[nt][r] * inv[r]);
            // out = P @ v  (M=16, K=64, N=32)
            v8f ov[2];
            for (int dt = 0; dt < 2; ++dt)
                for (int r = 0; r < 8; ++r) ov[dt][r] = 0.f;
            for (int kc = 0; kc < 2; ++kc) {
                v16bf pA = load_op(Pb + idx * 72 + kc * 32, h);
                for (int dt = 0; dt < 2; ++dt)
                    ov[dt] = WMMA_BF16(pA, vB[dt][kc], ov[dt]);
            }
            // attn-out -> xA region (reused), [t][head*32+d], stride 392
            for (int dt = 0; dt < 2; ++dt)
                for (int r = 0; r < 8; ++r) {
                    int t = mt * 16 + 8 * h + r;
                    sh[t * 392 + head * 32 + dt * 16 + idx] = (__bf16)(ov[dt][r]);
                }
        }
    }
    __syncthreads();

    // ---------------- phase 3: output projection + window reverse ----------------
    for (int i = 0; i < 2; ++i) {
        int o = (wave * 2 + i) * 16 + idx;
        const __bf16* wrow = wp + (long)o * 384;
        v8f acc[4];
        for (int m = 0; m < 4; ++m)
            for (int r = 0; r < 8; ++r) acc[m][r] = 0.f;
        for (int kc = 0; kc < 12; ++kc) {
            v16bf bop = load_op(wrow + kc * 32, h);
            for (int m = 0; m < 4; ++m) {
                v16bf aop = load_op(sh + (m * 16 + idx) * 392 + kc * 32, h);
                acc[m] = WMMA_BF16(aop, bop, acc[m]);
            }
        }
        float bias = proj_b[o];
        float* obase = out + (long)o * 50176 + pixbase;
        for (int m = 0; m < 4; ++m) {
            int ty = 2 * m + h;                    // 8 accum rows = 8 consecutive tx
            float* p = obase + ty * 224;
            v4f lo, hi;
            for (int r = 0; r < 4; ++r) {
                lo[r] = acc[m][r]     + bias;
                hi[r] = acc[m][r + 4] + bias;
            }
            *(v4f*)p       = lo;                   // two aligned 16B stores
            *((v4f*)p + 1) = hi;
        }
    }
}

extern "C" void kernel_launch(void* const* d_in, const int* in_sizes, int n_in,
                              void* d_out, int out_size, void* d_ws, size_t ws_size,
                              hipStream_t stream) {
    const float* x      = (const float*)d_in[0];
    const float* qkv_w  = (const float*)d_in[1];
    const float* qkv_b  = (const float*)d_in[2];
    const float* proj_w = (const float*)d_in[3];
    const float* proj_b = (const float*)d_in[4];
    const float* btab   = (const float*)d_in[5];
    float* outp = (float*)d_out;

    __bf16* wq = (__bf16*)d_ws;                       // 1152*384 bf16 = 884736 B
    __bf16* wp = (__bf16*)((char*)d_ws + 884736);     // 384*384 bf16  = 294912 B

    (void)in_sizes; (void)n_in; (void)out_size; (void)ws_size;

    hipFuncSetAttribute((const void*)win_attn_fused,
                        hipFuncAttributeMaxDynamicSharedMemorySize, LDS_BYTES);

    prep_weights<<<(1152 * 384 + 255) / 256, 256, 0, stream>>>(qkv_w, proj_w, wq, wp);
    win_attn_fused<<<3136, 384, LDS_BYTES, stream>>>(x, wq, qkv_b, wp, proj_b, btab, outp);
}
